// QWA_48661979464273
// MI455X (gfx1250) — matrix operand order repaired and checked by reading, not compiled, not run on hardware
//
#include <hip/hip_runtime.h>
#include <hip/hip_bf16.h>

// ---------------- problem constants (match reference) ----------------
#define BSZ 4096            // B
#define LDIM 8192           // L = N*D
#define HDIM 256            // H
#define NCH  8              // C
#define NEXP 3              // K experts
#define UQF  0.9f
#define LQF  0.1f

#define MTILE 32            // rows per GEMM block (bucket alignment)
#define KC    32            // K-chunk per WMMA (bf16 16x16x32)
#define NKCH  (LDIM / KC)   // 256 K-chunks
#define MAX_TILES ((BSZ + NEXP*(MTILE-1) + MTILE-1) / MTILE)   // 131

// ---------------- workspace layout (bytes) ----------------
#define OFF_UQ      0            // 8 f32
#define OFF_LQ      64           // 8 f32
#define OFF_GROUP   128          // 4096 i32
#define OFF_PERM    16512        // MAX_TILES*32 i32 (<= 4192)
#define OFF_META    33344        // 16 i32
#define OFF_V       33408        // NEXP*HDIM f32
#define OFF_CVAL    36544        // NEXP f32
#define OFF_LOGITS  36608        // 4096 f32
#define OFF_W1BF    52992        // bf16 pre-swizzled W1: NEXP*NKCH tiles of [HDIM][KC]
#define W1BF_BYTES  ((size_t)NEXP * NKCH * HDIM * KC * 2)      // 12.58 MB
#define TILE_BYTES  (HDIM * KC * 2)                            // 16 KB per K-chunk tile

typedef __bf16 v16bf __attribute__((ext_vector_type(16)));
typedef __bf16 v8bf  __attribute__((ext_vector_type(8)));
typedef float  v8f   __attribute__((ext_vector_type(8)));

// float -> bf16 round-to-nearest-even (bit-exact, no libcall)
static __device__ __forceinline__ unsigned short f2bf_bits(float f) {
    unsigned u = __float_as_uint(f);
    unsigned r = (u + 0x7FFFu + ((u >> 16) & 1u)) >> 16;
    return (unsigned short)r;
}
static __device__ __forceinline__ unsigned pack2bf(float lo, float hi) {
    return (unsigned)f2bf_bits(lo) | ((unsigned)f2bf_bits(hi) << 16);
}
static __device__ __forceinline__ float gelu_exact(float x) {
    return 0.5f * x * (1.0f + erff(x * 0.70710678118654752f));
}

// ============================================================
// K1: per-channel thresholds via order-statistic counting.
// grid = NCH blocks, 256 threads.
// ============================================================
__global__ void qwa_thresholds(const float* __restrict__ q,
                               const int* __restrict__ ch,
                               float* __restrict__ uq_th,
                               float* __restrict__ lq_th) {
    const int c = blockIdx.x;
    const int tid = threadIdx.x;
    __shared__ float qs[BSZ];
    __shared__ int red[256];
    __shared__ int s_iu, s_il;

    const float INF = __builtin_inff();
    for (int i = tid; i < BSZ; i += 256)
        qs[i] = (ch[i] == c) ? q[i] : INF;
    __syncthreads();

    int cnt = 0;
    for (int i = tid; i < BSZ; i += 256) cnt += (qs[i] < INF) ? 1 : 0;
    red[tid] = cnt;
    __syncthreads();
    for (int st = 128; st > 0; st >>= 1) {
        if (tid < st) red[tid] += red[tid + st];
        __syncthreads();
    }
    if (tid == 0) {
        float n = (float)red[0];
        float au = ceilf((n + 1.0f) * UQF) / n;
        float pu = (au > 1.0f) ? UQF : au;
        float al = floorf((n + 1.0f) * LQF) / n;
        float pl = (al < 0.0f) ? LQF : al;
        s_iu = (int)floorf(pu * (n - 1.0f));   // 'lower' interpolation index
        s_il = (int)floorf(pl * (n - 1.0f));
    }
    __syncthreads();
    const int iu = s_iu, il = s_il;

    for (int i = tid; i < BSZ; i += 256) {
        float x = qs[i];
        if (!(x < INF)) continue;
        int less = 0, eq = 0;
        for (int j = 0; j < BSZ; ++j) {
            float y = qs[j];
            less += (y < x) ? 1 : 0;
            eq   += (y == x) ? 1 : 0;
        }
        if (less <= iu && iu < less + eq) uq_th[c] = x;
        if (less <= il && il < less + eq) lq_th[c] = x;
    }
}

// ============================================================
// K2: group assignment + bucketed row permutation (padded to MTILE).
// grid = 1 block, 256 threads.
// ============================================================
__global__ void qwa_group_scatter(const float* __restrict__ q,
                                  const int* __restrict__ ch,
                                  const float* __restrict__ uq_th,
                                  const float* __restrict__ lq_th,
                                  int* __restrict__ group,
                                  int* __restrict__ perm,
                                  int* __restrict__ meta) {
    __shared__ int cnt[NEXP], cur[NEXP], base[NEXP], s_total;
    const int tid = threadIdx.x;
    if (tid < NEXP) { cnt[tid] = 0; cur[tid] = 0; }
    __syncthreads();

    for (int i = tid; i < BSZ; i += 256) {
        int c = ch[i];
        float x = q[i];
        int g = (x <= lq_th[c]) ? 2 : ((x >= uq_th[c]) ? 0 : 1);  // lower wins overlap
        group[i] = g;
        atomicAdd(&cnt[g], 1);
    }
    __syncthreads();
    if (tid == 0) {
        int p0 = ((cnt[0] + MTILE - 1) / MTILE) * MTILE;
        int p1 = ((cnt[1] + MTILE - 1) / MTILE) * MTILE;
        int p2 = ((cnt[2] + MTILE - 1) / MTILE) * MTILE;
        base[0] = 0; base[1] = p0; base[2] = p0 + p1;
        s_total = p0 + p1 + p2;
        meta[0] = s_total / MTILE;
    }
    __syncthreads();
    const int total = s_total;
    for (int i = tid; i < total; i += 256) perm[i] = -1;
    __syncthreads();
    for (int i = tid; i < BSZ; i += 256) {
        int g = group[i];
        int pos = base[g] + atomicAdd(&cur[g], 1);
        perm[pos] = i;
    }
}

// ============================================================
// K3: head folding  v[k] = W2[k] @ Wh[k],  c[k] = b2[k].Wh[k] + bh[k]
// grid = NEXP*(HDIM+1) blocks, 256 threads.
// ============================================================
__global__ void qwa_head_fold(const float* __restrict__ W2,
                              const float* __restrict__ b2,
                              const float* __restrict__ Wh,
                              const float* __restrict__ bh,
                              float* __restrict__ vvec,
                              float* __restrict__ cval) {
    const int k = blockIdx.x / (HDIM + 1);
    const int j = blockIdx.x % (HDIM + 1);
    const int tid = threadIdx.x;
    __shared__ float red[256];
    const float* wh = Wh + (size_t)k * LDIM;
    float s = 0.0f;
    if (j < HDIM) {
        const float* row = W2 + ((size_t)k * HDIM + j) * LDIM;
        for (int l = tid; l < LDIM; l += 256) s += row[l] * wh[l];
    } else {
        const float* row = b2 + (size_t)k * LDIM;
        for (int l = tid; l < LDIM; l += 256) s += row[l] * wh[l];
    }
    red[tid] = s;
    __syncthreads();
    for (int st = 128; st > 0; st >>= 1) {
        if (tid < st) red[tid] += red[tid + st];
        __syncthreads();
    }
    if (tid == 0) {
        if (j < HDIM) vvec[k * HDIM + j] = red[0];
        else          cval[k] = red[0] + bh[k];
    }
}

// ============================================================
// K3b: one-time W1 f32 -> bf16 pre-swizzle into per-K-chunk
// [col][k] tiles (byte-identical to the GEMM's LDS image).
// grid = NEXP*NKCH blocks, 256 threads (thread = column).
// ============================================================
__global__ void qwa_w1_swizzle(const float* __restrict__ W1,
                               unsigned* __restrict__ W1bf) {
    const int g     = blockIdx.x / NKCH;
    const int chunk = blockIdx.x % NKCH;
    const int n     = threadIdx.x;                 // column 0..255
    const int kk    = chunk * KC;
    const float* src = W1 + ((size_t)g * LDIM + kk) * HDIM + n;
    unsigned* dst = W1bf + ((size_t)blockIdx.x * HDIM * KC + n * KC) / 2;
    #pragma unroll
    for (int k2 = 0; k2 < KC; k2 += 2) {
        float f0 = src[(size_t)k2 * HDIM];
        float f1 = src[(size_t)(k2 + 1) * HDIM];
        dst[k2 / 2] = pack2bf(f0, f1);
    }
}

// ============================================================
// K4: fused gathered GEMM + GELU + folded-head dot -> logits.
// One block = 32 gathered rows (uniform expert g) x full H=256.
// 8 waves; wave w owns columns [w*32, w*32+32): 2M x 2N WMMA accumulators.
// Double-buffered LDS software pipeline: iteration computing buffer p
// stages K-chunk kk+KC into buffer 1-p first, so the async DMA flies
// under the WMMAs; ASYNCcnt/DScnt drained at the end-of-iteration barrier.
// FASTB=1: B tiles pre-swizzled bf16, global_load_async_to_lds_b128.
// FASTB=0: B converted f32->bf16 in-loop (fallback if ws too small).
// grid = MAX_TILES blocks, 256 threads.
// ============================================================
template <bool FASTB>
__global__ void __launch_bounds__(256)
qwa_expert_gemm(const float* __restrict__ z,
                const float* __restrict__ W1,
                const char* __restrict__ W1bf,
                const float* __restrict__ b1,
                const int* __restrict__ perm,
                const int* __restrict__ group,
                const int* __restrict__ meta,
                const float* __restrict__ vvec,
                const float* __restrict__ cval,
                float* __restrict__ logits) {
    if ((int)blockIdx.x >= meta[0]) return;

    __shared__ __bf16 ldsA[2][MTILE * KC];     // [buf][row][k]
    __shared__ __bf16 ldsB[2][HDIM * KC];      // [buf][col][k]
    __shared__ float  rowsum[MTILE];
    __shared__ int    srow[MTILE];
    __shared__ int    sg;

    const int tid  = threadIdx.x;
    const int lane = tid & 31;
    const int wv   = tid >> 5;
    const int r    = lane & 15;
    const int half = lane >> 4;
    const int tile = blockIdx.x;

    if (tid < MTILE) srow[tid] = perm[tile * MTILE + tid];
    if (tid < MTILE) rowsum[tid] = 0.0f;
    __syncthreads();
    if (tid == 0) sg = group[srow[0]];         // first row of a tile is always valid
    __syncthreads();
    const int g = sg;
    const float* __restrict__ W1g   = W1 + (size_t)g * LDIM * HDIM;
    const char*  __restrict__ W1bfg = W1bf + (size_t)g * NKCH * TILE_BYTES;

    const int a_row = tid >> 3;                // 0..31
    const int a_c4  = (tid & 7) * 4;           // 0..28 step 4
    const int a_p   = srow[a_row];

    auto stageA = [&](int kk, int buf) {
        float4 f = make_float4(0.f, 0.f, 0.f, 0.f);
        if (a_p >= 0)
            f = *(const float4*)(z + (size_t)a_p * LDIM + kk + a_c4);
        uint2 pk;
        pk.x = pack2bf(f.x, f.y);
        pk.y = pack2bf(f.z, f.w);
        *(uint2*)&ldsA[buf][a_row * KC + a_c4] = pk;
    };
    auto stageB = [&](int kk, int buf) {
        if (FASTB) {
            // pre-swizzled bf16 tile: async DMA 16KB global -> LDS, no VALU
            const char* gsrc = W1bfg + (size_t)(kk >> 5) * TILE_BYTES + tid * 64;
            __builtin_prefetch(gsrc + TILE_BYTES, 0, 0);       // next K-chunk tile
            const unsigned lbase =
                (unsigned)(size_t)(&ldsB[buf][0]) + (unsigned)(tid * 64);
            #pragma unroll
            for (int j = 0; j < 4; ++j) {
                unsigned long long ga = (unsigned long long)(gsrc + j * 16);
                unsigned la = lbase + (unsigned)(j * 16);
                asm volatile("global_load_async_to_lds_b128 %0, %1, off"
                             :: "v"(la), "v"(ga) : "memory");
            }
        } else {
            const float* wcol = W1g + (size_t)kk * HDIM + tid;  // column = tid
            __builtin_prefetch(wcol + (size_t)KC * HDIM, 0, 0);
            #pragma unroll
            for (int k2 = 0; k2 < KC; k2 += 2) {
                float f0 = wcol[(size_t)k2 * HDIM];
                float f1 = wcol[(size_t)(k2 + 1) * HDIM];
                *(unsigned*)&ldsB[buf][tid * KC + k2] = pack2bf(f0, f1);
            }
        }
    };

    v8f acc[2][2];
    acc[0][0] = (v8f){}; acc[0][1] = (v8f){};
    acc[1][0] = (v8f){}; acc[1][1] = (v8f){};

    // ---- pipeline prologue: fill buffer 0 ----
    stageA(0, 0);
    stageB(0, 0);
    if (FASTB) asm volatile("s_wait_asynccnt 0x0" ::: "memory");
    __syncthreads();

    for (int kk = 0; kk < LDIM; kk += KC) {
        const int p = (kk >> 5) & 1;
        // ---- stage next K-chunk into the idle buffer (overlaps compute) ----
        if (kk + KC < LDIM) {
            stageA(kk + KC, 1 - p);
            stageB(kk + KC, 1 - p);
        }
        // ---- fragments per ISA 16-bit A(16x32)/B(32x16) lane striping ----
        v16bf afrag[2], bfrag[2];
        #pragma unroll
        for (int m = 0; m < 2; ++m) {
            const v8bf lo = *(const v8bf*)&ldsA[p][(m * 16 + r) * KC + half * 8];
            const v8bf hi = *(const v8bf*)&ldsA[p][(m * 16 + r) * KC + 16 + half * 8];
            afrag[m] = __builtin_shufflevector(lo, hi,
                0,1,2,3,4,5,6,7,8,9,10,11,12,13,14,15);
        }
        #pragma unroll
        for (int n = 0; n < 2; ++n) {
            const int col = wv * 32 + n * 16 + r;
            const v8bf lo = *(const v8bf*)&ldsB[p][col * KC + half * 8];
            const v8bf hi = *(const v8bf*)&ldsB[p][col * KC + 16 + half * 8];
            bfrag[n] = __builtin_shufflevector(lo, hi,
                0,1,2,3,4,5,6,7,8,9,10,11,12,13,14,15);
        }
        #pragma unroll
        for (int m = 0; m < 2; ++m)
            #pragma unroll
            for (int n = 0; n < 2; ++n)
                acc[m][n] = __builtin_amdgcn_wmma_f32_16x16x32_bf16(
                    false, afrag[m], false, bfrag[n],
                    (short)0, acc[m][n], false, false);
        // ---- drain this wave's async issues, then swap buffers ----
        if (FASTB && (kk + KC < LDIM))
            asm volatile("s_wait_asynccnt 0x0" ::: "memory");
        __syncthreads();   // DScnt drained + barrier: buf 1-p complete, buf p free
    }

    // ---- epilogue: +b1, exact GELU, dot with folded head v[g] ----
    #pragma unroll
    for (int n = 0; n < 2; ++n) {
        const int h  = wv * 32 + n * 16 + r;
        const float vg = vvec[g * HDIM + h];
        const float bg = b1[g * HDIM + h];
        #pragma unroll
        for (int m = 0; m < 2; ++m) {
            #pragma unroll
            for (int i = 0; i < 8; ++i) {
                const int row = m * 16 + i + 8 * half;     // C/D VGPR layout
                float x = acc[m][n][i] + bg;
                atomicAdd(&rowsum[row], gelu_exact(x) * vg);
            }
        }
    }
    __syncthreads();
    if (tid < MTILE) {
        int p = srow[tid];
        if (p >= 0) logits[p] = rowsum[tid] + cval[g];
    }
}

// ============================================================
// K5: per-channel stable softmax, out = q * softmax_ch(logits)
// grid = NCH blocks, 256 threads.
// ============================================================
__global__ void qwa_softmax(const float* __restrict__ q,
                            const int* __restrict__ ch,
                            const float* __restrict__ logits,
                            float* __restrict__ out) {
    const int c = blockIdx.x;
    const int tid = threadIdx.x;
    __shared__ float red[256];
    __shared__ float s_m, s_s;

    float m = -__builtin_inff();
    for (int i = tid; i < BSZ; i += 256)
        if (ch[i] == c) m = fmaxf(m, logits[i]);
    red[tid] = m;
    __syncthreads();
    for (int st = 128; st > 0; st >>= 1) {
        if (tid < st) red[tid] = fmaxf(red[tid], red[tid + st]);
        __syncthreads();
    }
    if (tid == 0) s_m = red[0];
    __syncthreads();
    const float mm = s_m;

    float s = 0.0f;
    for (int i = tid; i < BSZ; i += 256)
        if (ch[i] == c) s += expf(logits[i] - mm);
    red[tid] = s;
    __syncthreads();
    for (int st = 128; st > 0; st >>= 1) {
        if (tid < st) red[tid] += red[tid + st];
        __syncthreads();
    }
    if (tid == 0) s_s = red[0];
    __syncthreads();
    const float inv = 1.0f / s_s;

    for (int i = tid; i < BSZ; i += 256)
        if (ch[i] == c) out[i] = q[i] * expf(logits[i] - mm) * inv;
}

// ============================================================
extern "C" void kernel_launch(void* const* d_in, const int* in_sizes, int n_in,
                              void* d_out, int out_size, void* d_ws, size_t ws_size,
                              hipStream_t stream) {
    const float* z    = (const float*)d_in[0];
    const float* q    = (const float*)d_in[1];
    const int*   chid = (const int*)  d_in[2];
    const float* W1   = (const float*)d_in[3];
    const float* b1   = (const float*)d_in[4];
    const float* W2   = (const float*)d_in[5];
    const float* b2   = (const float*)d_in[6];
    const float* Wh   = (const float*)d_in[7];
    const float* bh   = (const float*)d_in[8];
    float* out = (float*)d_out;

    char* ws = (char*)d_ws;
    float* uq_th  = (float*)(ws + OFF_UQ);
    float* lq_th  = (float*)(ws + OFF_LQ);
    int*   group  = (int*)  (ws + OFF_GROUP);
    int*   perm   = (int*)  (ws + OFF_PERM);
    int*   meta   = (int*)  (ws + OFF_META);
    float* vvec   = (float*)(ws + OFF_V);
    float* cval   = (float*)(ws + OFF_CVAL);
    float* logits = (float*)(ws + OFF_LOGITS);
    char*  W1bf   = ws + OFF_W1BF;

    const bool fast = (ws_size >= (size_t)OFF_W1BF + W1BF_BYTES);

    qwa_thresholds<<<NCH, 256, 0, stream>>>(q, chid, uq_th, lq_th);
    qwa_group_scatter<<<1, 256, 0, stream>>>(q, chid, uq_th, lq_th,
                                             group, perm, meta);
    qwa_head_fold<<<NEXP * (HDIM + 1), 256, 0, stream>>>(W2, b2, Wh, bh,
                                                         vvec, cval);
    if (fast) {
        qwa_w1_swizzle<<<NEXP * NKCH, 256, 0, stream>>>(W1, (unsigned*)W1bf);
        qwa_expert_gemm<true><<<MAX_TILES, 256, 0, stream>>>(
            z, W1, W1bf, b1, perm, group, meta, vvec, cval, logits);
    } else {
        qwa_expert_gemm<false><<<MAX_TILES, 256, 0, stream>>>(
            z, W1, W1bf, b1, perm, group, meta, vvec, cval, logits);
    }
    qwa_softmax<<<NCH, 256, 0, stream>>>(q, chid, logits, out);
}
